// DNDF_6828998000820
// MI455X (gfx1250) — matrix operand order repaired and checked by reading, not compiled, not run on hardware
//
#include <hip/hip_runtime.h>
#include <hip/hip_bf16.h>

// ---------------------------------------------------------------------------
// DNDF forward for MI455X (gfx1250, wave32, WMMA bf16 16x16x32, f32 accum)
//
//   GEMM1: tree_net = x(2048x1024) @ W^T(1024x8064)        (bf16 WMMA)
//   route: sigmoid + 6-factor path products -> features     (VALU + LDS)
//   GEMM2: out = features(2048x8192) @ softmax(leaves)      (bf16 WMMA)
// ---------------------------------------------------------------------------

typedef __attribute__((ext_vector_type(16))) __bf16 bf16x16;
typedef __attribute__((ext_vector_type(8)))  __bf16 bf16x8;
typedef __attribute__((ext_vector_type(8)))  float  f32x8;

#define KP 40   // LDS row stride in bf16 elements (80 bytes: 16B aligned, bank-spread)

// round-to-nearest-even fp32 -> bf16 (pure integer ops: always legal)
__device__ __forceinline__ __bf16 f2bf(float f) {
    union { float f; unsigned u; } v; v.f = f;
    unsigned r = v.u + 0x7FFFu + ((v.u >> 16) & 1u);
    unsigned short h = (unsigned short)(r >> 16);
    return __builtin_bit_cast(__bf16, h);
}

// A fragment (16x32 bf16, M x K): lane m holds K = {off..off+7, 16+off..16+off+7},
// off = 8*(lane>=16).  p points at LDS row 0 of the 16-row subtile (row stride KP).
__device__ __forceinline__ bf16x16 ldFragA(const __bf16* p, int lane) {
    const int m   = lane & 15;
    const int off = (lane >> 4) * 8;
    const __bf16* q = p + m * KP + off;
    bf16x8 lo = *(const bf16x8*)(q);
    bf16x8 hi = *(const bf16x8*)(q + 16);
    return __builtin_shufflevector(lo, hi, 0,1,2,3,4,5,6,7,8,9,10,11,12,13,14,15);
}

// B fragment (32x16 bf16, K x N): lane n holds column n, K = {off..off+7, 16+off..16+off+7}, off = 8*(lane>=16).
// LDS tile is stored [n][k] (column n of B contiguous in k), row stride KP.
__device__ __forceinline__ bf16x16 ldFragB(const __bf16* p, int lane) {
    const int n  = lane & 15;
    const int kh = (lane >> 4) * 8;
    const __bf16* q = p + n * KP + kh;
    bf16x8 lo = *(const bf16x8*)(q);
    bf16x8 hi = *(const bf16x8*)(q + 16);
    return __builtin_shufflevector(lo, hi, 0,1,2,3,4,5,6,7,8,9,10,11,12,13,14,15);
}

// ---------------------------------------------------------------------------
// fp32 -> bf16 bulk convert
// ---------------------------------------------------------------------------
__global__ __launch_bounds__(256) void cvt_bf16_kernel(
    const float* __restrict__ src, __bf16* __restrict__ dst, int n)
{
    int i = blockIdx.x * 256 + threadIdx.x;
    int stride = gridDim.x * 256;
    for (; i < n; i += stride) dst[i] = f2bf(src[i]);
}

// ---------------------------------------------------------------------------
// row softmax of leaves (8192 x 512) -> bf16
// ---------------------------------------------------------------------------
__global__ __launch_bounds__(256) void softmax_kernel(
    const float* __restrict__ L, __bf16* __restrict__ S)
{
    __shared__ float red[256];
    const int r   = blockIdx.x;
    const int tid = threadIdx.x;
    const float* row = L + (size_t)r * 512;
    float v0 = row[tid], v1 = row[tid + 256];

    red[tid] = fmaxf(v0, v1);
    __syncthreads();
    for (int s = 128; s > 0; s >>= 1) {
        if (tid < s) red[tid] = fmaxf(red[tid], red[tid + s]);
        __syncthreads();
    }
    float M = red[0];
    __syncthreads();

    float e0 = __expf(v0 - M), e1 = __expf(v1 - M);
    red[tid] = e0 + e1;
    __syncthreads();
    for (int s = 128; s > 0; s >>= 1) {
        if (tid < s) red[tid] += red[tid + s];
        __syncthreads();
    }
    float inv = 1.0f / red[0];
    S[(size_t)r * 512 + tid]       = f2bf(e0 * inv);
    S[(size_t)r * 512 + tid + 256] = f2bf(e1 * inv);
}

// ---------------------------------------------------------------------------
// GEMM1: C(2048x8064 f32) = A(2048x1024 bf16) * W^T   (W: 8064x1024 bf16 row-major)
// Block tile 128x128, BK=32, 256 threads (8 waves), wave tile 32x64 (2x4 WMMA).
// B[k][n] = W[n][k] -> W rows copied straight into the [n][k] LDS layout.
// ---------------------------------------------------------------------------
__global__ __launch_bounds__(256) void gemm1_wmma(
    const __bf16* __restrict__ A,
    const __bf16* __restrict__ W,
    float* __restrict__ C)
{
    __shared__ __bf16 ldsA[2][128][KP];
    __shared__ __bf16 ldsB[2][128][KP];

    const int tid   = threadIdx.x;
    const int lane  = tid & 31;
    const int wid   = tid >> 5;
    const int waveM = (wid & 3) * 32;
    const int waveN = (wid >> 2) * 64;
    const int tileN = blockIdx.x * 128;
    const int tileM = blockIdx.y * 128;

    f32x8 acc[2][4];
    for (int i = 0; i < 2; ++i)
        for (int j = 0; j < 4; ++j)
            for (int r = 0; r < 8; ++r) acc[i][j][r] = 0.0f;

    // global->reg staging: 128x32 bf16 tile = 512 blocks of 8; 2 per thread
    int rr[2], cc[2];
    for (int ph = 0; ph < 2; ++ph) {
        int bb = tid + 256 * ph;
        rr[ph] = bb >> 2;
        cc[ph] = (bb & 3) * 8;
    }
    bf16x8 rA[2], rB[2];

    auto gload = [&](int kt) {
        for (int ph = 0; ph < 2; ++ph) {
            rA[ph] = *(const bf16x8*)(A + (size_t)(tileM + rr[ph]) * 1024 + kt * 32 + cc[ph]);
            rB[ph] = *(const bf16x8*)(W + (size_t)(tileN + rr[ph]) * 1024 + kt * 32 + cc[ph]);
        }
    };
    auto lstore = [&](int buf) {
        for (int ph = 0; ph < 2; ++ph) {
            *(bf16x8*)&ldsA[buf][rr[ph]][cc[ph]] = rA[ph];
            *(bf16x8*)&ldsB[buf][rr[ph]][cc[ph]] = rB[ph];
        }
    };

    gload(0);
    lstore(0);

    const int KT = 1024 / 32;
    for (int kt = 0; kt < KT; ++kt) {
        const int cur = kt & 1;
        if (kt + 1 < KT) gload(kt + 1);
        __syncthreads();

        bf16x16 af[2], bfr[4];
        for (int i = 0; i < 2; ++i) af[i]  = ldFragA(&ldsA[cur][waveM + i * 16][0], lane);
        for (int j = 0; j < 4; ++j) bfr[j] = ldFragB(&ldsB[cur][waveN + j * 16][0], lane);
        for (int i = 0; i < 2; ++i)
            for (int j = 0; j < 4; ++j)
                acc[i][j] = __builtin_amdgcn_wmma_f32_16x16x32_bf16(
                    false, af[i], false, bfr[j], (short)0, acc[i][j], false, false);

        if (kt + 1 < KT) lstore((kt + 1) & 1);
    }

    // epilogue: C/D layout — VGPR r: M = r + 8*(lane>=16), N = lane&15
    const int mlo = (lane >> 4) * 8;
    const int nl  = lane & 15;
    for (int i = 0; i < 2; ++i)
        for (int j = 0; j < 4; ++j)
            for (int r = 0; r < 8; ++r) {
                int row = tileM + waveM + i * 16 + mlo + r;
                int col = tileN + waveN + j * 16 + nl;
                C[(size_t)row * 8064 + col] = acc[i][j][r];
            }
}

// ---------------------------------------------------------------------------
// routing: sigmoid(tree_net + b) -> 64 leaf path products per tree -> bf16 features
// one block per batch row; sigmoids staged in LDS (8064 f32 = 32 KB)
// ---------------------------------------------------------------------------
__global__ __launch_bounds__(256) void routes_kernel(
    const float* __restrict__ tn,   // 2048 x 8064
    const float* __restrict__ bias, // 8064
    __bf16* __restrict__ F)         // 2048 x 8192
{
    __shared__ float sp[8064];
    const int b   = blockIdx.x;
    const int tid = threadIdx.x;

    for (int i = tid; i < 8064; i += 256) {
        float v = tn[(size_t)b * 8064 + i] + bias[i];
        sp[i] = 1.0f / (1.0f + __expf(-v));
    }
    __syncthreads();

    for (int j = tid; j < 8192; j += 256) {
        int t = j >> 6, leaf = j & 63;
        const float* ps = sp + t * 63;
        float prod = 1.0f;
#pragma unroll
        for (int d = 0; d < 6; ++d) {
            int node = (1 << d) - 1 + (leaf >> (6 - d));
            int dir  = (leaf >> (5 - d)) & 1;
            float s  = ps[node];
            prod *= dir ? (1.0f - s) : s;
        }
        F[(size_t)b * 8192 + j] = f2bf(prod);
    }
}

// ---------------------------------------------------------------------------
// GEMM2: out(2048x512 f32) = F(2048x8192 bf16) * S(8192x512 bf16), scaled 1/128
// Block tile 64x64, BK=32, 128 threads (4 waves), wave tile 32x32 (2x2 WMMA).
// S is row-major -> transposed into the [n][k] LDS layout on store.
// ---------------------------------------------------------------------------
__global__ __launch_bounds__(128) void gemm2_wmma(
    const __bf16* __restrict__ F,
    const __bf16* __restrict__ S,
    float* __restrict__ out)
{
    __shared__ __bf16 ldsA[2][64][KP];
    __shared__ __bf16 ldsB[2][64][KP];

    const int tid   = threadIdx.x;
    const int lane  = tid & 31;
    const int wid   = tid >> 5;
    const int waveM = (wid & 1) * 32;
    const int waveN = (wid >> 1) * 32;
    const int tileN = blockIdx.x * 64;
    const int tileM = blockIdx.y * 64;

    f32x8 acc[2][2];
    for (int i = 0; i < 2; ++i)
        for (int j = 0; j < 2; ++j)
            for (int r = 0; r < 8; ++r) acc[i][j][r] = 0.0f;

    // A: 64x32 tile, 256 8-elem blocks, 2/thread.  B: 32(k)x64(n), 256 blocks, 2/thread.
    int ar[2], ac[2], bk[2], bn[2];
    for (int ph = 0; ph < 2; ++ph) {
        int bb = tid + 128 * ph;
        ar[ph] = bb >> 2;
        ac[ph] = (bb & 3) * 8;
        bk[ph] = bb >> 3;
        bn[ph] = bb & 7;
    }
    bf16x8 rA[2], rB[2];

    auto gload = [&](int kt) {
        for (int ph = 0; ph < 2; ++ph) {
            rA[ph] = *(const bf16x8*)(F + (size_t)(tileM + ar[ph]) * 8192 + kt * 32 + ac[ph]);
            rB[ph] = *(const bf16x8*)(S + (size_t)(kt * 32 + bk[ph]) * 512 + tileN + bn[ph] * 8);
        }
    };
    auto lstore = [&](int buf) {
        for (int ph = 0; ph < 2; ++ph) {
            *(bf16x8*)&ldsA[buf][ar[ph]][ac[ph]] = rA[ph];
#pragma unroll
            for (int j = 0; j < 8; ++j)          // transpose: LDS[n][k] = S[k][n]
                ldsB[buf][bn[ph] * 8 + j][bk[ph]] = rB[ph][j];
        }
    };

    gload(0);
    lstore(0);

    const int KT = 8192 / 32;
    for (int kt = 0; kt < KT; ++kt) {
        const int cur = kt & 1;
        if (kt + 1 < KT) gload(kt + 1);
        __syncthreads();

        bf16x16 af[2], bfr[2];
        for (int i = 0; i < 2; ++i) af[i]  = ldFragA(&ldsA[cur][waveM + i * 16][0], lane);
        for (int j = 0; j < 2; ++j) bfr[j] = ldFragB(&ldsB[cur][waveN + j * 16][0], lane);
        for (int i = 0; i < 2; ++i)
            for (int j = 0; j < 2; ++j)
                acc[i][j] = __builtin_amdgcn_wmma_f32_16x16x32_bf16(
                    false, af[i], false, bfr[j], (short)0, acc[i][j], false, false);

        if (kt + 1 < KT) lstore((kt + 1) & 1);
    }

    const int mlo = (lane >> 4) * 8;
    const int nl  = lane & 15;
    for (int i = 0; i < 2; ++i)
        for (int j = 0; j < 2; ++j)
            for (int r = 0; r < 8; ++r) {
                int row = tileM + waveM + i * 16 + mlo + r;
                int col = tileN + waveN + j * 16 + nl;
                out[(size_t)row * 512 + col] = acc[i][j][r] * 0.0078125f; // /128
            }
}

// ---------------------------------------------------------------------------
// launch
// ---------------------------------------------------------------------------
extern "C" void kernel_launch(void* const* d_in, const int* in_sizes, int n_in,
                              void* d_out, int out_size, void* d_ws, size_t ws_size,
                              hipStream_t stream) {
    (void)in_sizes; (void)n_in; (void)out_size; (void)ws_size;

    const float* x      = (const float*)d_in[0];  // 2048 x 1024
    const float* W      = (const float*)d_in[1];  // 8064 x 1024
    const float* bias   = (const float*)d_in[2];  // 8064
    const float* leaves = (const float*)d_in[3];  // 8192 x 512
    float* out = (float*)d_out;                   // 2048 x 512

    char* ws = (char*)d_ws;
    __bf16* xb = (__bf16*)(ws);                               //  4,194,304 B
    __bf16* Wb = (__bf16*)(ws + 4194304);                     // 16,515,072 B
    __bf16* Sb = (__bf16*)(ws + 4194304 + 16515072);          //  8,388,608 B
    float*  tn = (float*)(ws + 4194304 + 16515072 + 8388608); // 66,060,288 B
    __bf16* Fb = (__bf16*)(ws + 4194304 + 16515072 + 8388608 + 66060288); // 33,554,432 B

    cvt_bf16_kernel<<<1024, 256, 0, stream>>>(x, xb, 2048 * 1024);
    cvt_bf16_kernel<<<2048, 256, 0, stream>>>(W, Wb, 8064 * 1024);
    softmax_kernel<<<8192, 256, 0, stream>>>(leaves, Sb);
    gemm1_wmma<<<dim3(63, 16), 256, 0, stream>>>(xb, Wb, tn);
    routes_kernel<<<2048, 256, 0, stream>>>(tn, bias, Fb);
    gemm2_wmma<<<dim3(8, 32), 128, 0, stream>>>(Fb, Sb, out);
}
